// MyMultiheadAttention_36009005809779
// MI455X (gfx1250) — matrix operand order repaired and checked by reading, not compiled, not run on hardware
//
#include <hip/hip_runtime.h>
#include <math.h>

// ---------------------------------------------------------------------------
// MHA forward for MI455X (gfx1250, wave32, WMMA bf16 + TDM async staging)
// B=4, S=2048, E=1024, H=16, D=64
// ---------------------------------------------------------------------------

constexpr int kB = 4;
constexpr int kS = 2048;
constexpr int kE = 1024;
constexpr int kH = 16;
constexpr int kD = 64;
constexpr int kM = kB * kS;  // 8192 rows of x viewed as a matrix

typedef __attribute__((ext_vector_type(16))) __bf16        bf16x16;
typedef __attribute__((ext_vector_type(4)))  __bf16        bf16x4;
typedef __attribute__((ext_vector_type(8)))  float         f32x8;
typedef __attribute__((ext_vector_type(4)))  float         f32x4;
typedef __attribute__((ext_vector_type(4)))  unsigned int  u32x4;
typedef __attribute__((ext_vector_type(4)))  unsigned int  v4u;
typedef __attribute__((ext_vector_type(8)))  int           v8i;
typedef __attribute__((ext_vector_type(4)))  int           v4i;

// WMMA fragment: 16 bf16 values per lane = 32 bytes = two 16-byte chunks.
union Frag {
  bf16x16 v;
  u32x4   u[2];
};

__device__ __forceinline__ f32x8 wmma_bf16f32(const Frag& a, const Frag& b, f32x8 c) {
  // D = A(16x32 bf16) * B(32x16 bf16) + C(16x16 f32)
  return __builtin_amdgcn_wmma_f32_16x16x32_bf16(
      /*neg_a=*/false, a.v, /*neg_b=*/false, b.v,
      /*c_mod=*/(short)0, c, /*reuse_a=*/false, /*reuse_b=*/false);
}

__device__ __forceinline__ f32x8 fzero8() {
  f32x8 z;
#pragma unroll
  for (int i = 0; i < 8; ++i) z[i] = 0.0f;
  return z;
}

// ---------------------------------------------------------------------------
// TDM: async 2D tile load (global -> LDS) per CDNA5 ISA.
// Tile is 64 bf16 wide x tile_h rows from a row-major tensor whose rows are
// row_stride elements long. LDS destination rows are padded by hardware:
// pad_enable=1, pad_interval=4 (every 32 DWORDs = 128B = one 64-elem row),
// pad_amount=3 (4 DWORDs = 16B = 8 bf16)  ->  LDS row stride 72 bf16.
// Issued by one wave (EXEC ignored); tracked with TENSORcnt.
// ---------------------------------------------------------------------------
__device__ __forceinline__ void tdm_load_tile64(const __bf16* gbase,
                                                __bf16* lds,
                                                unsigned tile_h,
                                                unsigned row_stride) {
  unsigned long long ga = (unsigned long long)(uintptr_t)gbase;
  unsigned lds_off = (unsigned)(uintptr_t)lds;  // low 32 bits = LDS offset
  v4u g0;
  g0[0] = 1u;                                  // count=1 (valid user D#)
  g0[1] = lds_off;                             // lds_addr
  g0[2] = (unsigned)(ga & 0xffffffffu);        // global_addr[31:0]
  g0[3] = (unsigned)((ga >> 32) & 0x01ffffffu) // global_addr[56:32]
          | 0x80000000u;                       // type=2 ("image") at [127:126]
  unsigned td0 = row_stride;                   // tensor_dim0 (elements)
  unsigned td1 = tile_h;                       // tensor_dim1 (rows present)
  v8i g1;
  g1[0] = (int)((1u << 16)      // data_size=1 -> 2 bytes
                | (1u << 20)    // pad_enable
                | (4u << 22)    // pad_interval: 32 DWORDs
                | (3u << 25));  // pad_amount: 4 DWORDs
  g1[1] = (int)(td0 << 16);                    // tensor_dim0[15:0]
  g1[2] = (int)((td0 >> 16) | (td1 << 16));    // tensor_dim0[31:16], dim1[15:0]
  g1[3] = (int)((td1 >> 16) | (64u << 16));    // dim1[31:16], tile_dim0=64
  g1[4] = (int)(tile_h & 0xffffu);             // tile_dim1; tile_dim2=0
  g1[5] = (int)row_stride;                     // tensor_dim0_stride[31:0]
  g1[6] = 0;                                   // stride0[47:32], dim1_stride lo
  g1[7] = 0;
  v4i z4 = {0, 0, 0, 0};
#if defined(__clang_major__) && (__clang_major__ >= 23)
  v8i z8 = {0, 0, 0, 0, 0, 0, 0, 0};
  __builtin_amdgcn_tensor_load_to_lds(g0, g1, z4, z4, z8, 0);
#else
  __builtin_amdgcn_tensor_load_to_lds(g0, g1, z4, z4, 0);
#endif
}

// ---------------------------------------------------------------------------
// 16-lane butterfly reductions in pure VALU via DPP16 (no LDS traffic).
// ---------------------------------------------------------------------------
template <int CTRL>
__device__ __forceinline__ float dpp_bfly(float v) {
  int x = __float_as_int(v);
  return __int_as_float(
      __builtin_amdgcn_update_dpp(x, x, CTRL, 0xf, 0xf, true));
}
__device__ __forceinline__ float red_max16(float v) {
  v = fmaxf(v, dpp_bfly<0xB1>(v));   // xor1: quad_perm[1,0,3,2]
  v = fmaxf(v, dpp_bfly<0x4E>(v));   // xor2: quad_perm[2,3,0,1]
  v = fmaxf(v, dpp_bfly<0x141>(v));  // row_half_mirror
  v = fmaxf(v, dpp_bfly<0x140>(v));  // row_mirror
  return v;
}
__device__ __forceinline__ float red_sum16(float v) {
  v += dpp_bfly<0xB1>(v);
  v += dpp_bfly<0x4E>(v);
  v += dpp_bfly<0x141>(v);
  v += dpp_bfly<0x140>(v);
  return v;
}

// ---------------------------------------------------------------------------
// fp32 -> bf16 cast with optional scale folded in before rounding.
// ---------------------------------------------------------------------------
__global__ __launch_bounds__(256) void cast_f32_to_bf16(
    const float* __restrict__ src, __bf16* __restrict__ dst, int n, float scale) {
  int i = (blockIdx.x * 256 + threadIdx.x) * 4;
  if (i < n) {
    f32x4 f = *(const f32x4*)(src + i);
    bf16x4 o;
    o.x = (__bf16)(f.x * scale);
    o.y = (__bf16)(f.y * scale);
    o.z = (__bf16)(f.z * scale);
    o.w = (__bf16)(f.w * scale);
    *(bf16x4*)(dst + i) = o;
  }
}

// ---------------------------------------------------------------------------
// C[m,n] = sum_k A[m,k] * W[n,k]   (A @ W^T, matching F.linear)
// Block 256 threads (8 waves), block tile 128x128, wave tile 32x64.
// K staged via double-buffered TDM tiles of 128x64; wave 0 drives the DMA.
// ---------------------------------------------------------------------------
template <bool OUT_BF16>
__global__ __launch_bounds__(256) void gemm_xwT(
    const __bf16* __restrict__ A, const __bf16* __restrict__ W,
    __bf16* __restrict__ outb, float* __restrict__ outf,
    const float* __restrict__ bias, int Kdim, int Ndim) {
  __shared__ __align__(16) __bf16 sA[2][128 * 72];  // padded stride 72 (by TDM)
  __shared__ __align__(16) __bf16 sW[2][128 * 72];

  const int t    = threadIdx.x;
  const int lane = t & 31;
  const int wid  = t >> 5;
  const int nl   = lane & 15;
  const int hf   = lane >> 4;
  const int wm   = (wid & 3) * 32;   // wave M offset (4 waves)
  const int wn   = (wid >> 2) * 64;  // wave N offset (2 waves)
  const int bm   = blockIdx.y * 128;
  const int bn   = blockIdx.x * 128;

  f32x8 acc[2][4];
#pragma unroll
  for (int mi = 0; mi < 2; ++mi)
#pragma unroll
    for (int nj = 0; nj < 4; ++nj) acc[mi][nj] = fzero8();

  const int nk = Kdim / 64;
  if (wid == 0) {  // prologue: tile 0 into buffer 0
    tdm_load_tile64(&A[(size_t)bm * Kdim], &sA[0][0], 128, (unsigned)Kdim);
    tdm_load_tile64(&W[(size_t)bn * Kdim], &sW[0][0], 128, (unsigned)Kdim);
  }

  for (int it = 0; it < nk; ++it) {
    const int cur = it & 1;
    __syncthreads();  // all waves done computing on buffer cur^1 (tile it-1)
    if (wid == 0) {
      if (it + 1 < nk) {
        int k1 = (it + 1) * 64;
        tdm_load_tile64(&A[(size_t)bm * Kdim + k1], &sA[cur ^ 1][0], 128,
                        (unsigned)Kdim);
        tdm_load_tile64(&W[(size_t)bn * Kdim + k1], &sW[cur ^ 1][0], 128,
                        (unsigned)Kdim);
        __builtin_amdgcn_s_wait_tensorcnt(2);  // tile it complete (in-order)
      } else {
        __builtin_amdgcn_s_wait_tensorcnt(0);
      }
    }
    __syncthreads();  // tile it visible to all waves

    const __bf16* cA = &sA[cur][0];
    const __bf16* cW = &sW[cur][0];
#pragma unroll
    for (int kk = 0; kk < 64; kk += 32) {
      Frag a[2], b[4];
#pragma unroll
      for (int mi = 0; mi < 2; ++mi) {
        const __bf16* p = &cA[(wm + mi * 16 + nl) * 72 + kk + hf * 8];
        a[mi].u[0] = *(const u32x4*)p;
        a[mi].u[1] = *(const u32x4*)(p + 16);
      }
#pragma unroll
      for (int nj = 0; nj < 4; ++nj) {
        const __bf16* p = &cW[(wn + nj * 16 + nl) * 72 + kk + hf * 8];
        b[nj].u[0] = *(const u32x4*)p;
        b[nj].u[1] = *(const u32x4*)(p + 16);
      }
#pragma unroll
      for (int mi = 0; mi < 2; ++mi)
#pragma unroll
        for (int nj = 0; nj < 4; ++nj)
          acc[mi][nj] = wmma_bf16f32(a[mi], b[nj], acc[mi][nj]);
    }
  }

  // C/D layout: VGPR i, half hf -> row = i + 8*hf, col = nl.
#pragma unroll
  for (int mi = 0; mi < 2; ++mi)
#pragma unroll
    for (int nj = 0; nj < 4; ++nj)
#pragma unroll
      for (int i = 0; i < 8; ++i) {
        int row = bm + wm + mi * 16 + hf * 8 + i;
        int col = bn + wn + nj * 16 + nl;
        float v = acc[mi][nj][i];
        if (OUT_BF16) {
          outb[(size_t)row * Ndim + col] = (__bf16)v;
        } else {
          outf[(size_t)row * Ndim + col] = v + bias[col];
        }
      }
}

// ---------------------------------------------------------------------------
// Flash attention: one (batch, head, 64-query-row tile) per 128-thread block.
// K chunks (64 keys x 64 d) via double-buffered TDM; V staged manually
// (transposed); softmax in exp2 domain (scale folded into Wq cast).
// ---------------------------------------------------------------------------
__global__ __launch_bounds__(128) void flash_attn(
    const __bf16* __restrict__ Q, const __bf16* __restrict__ K,
    const __bf16* __restrict__ V, __bf16* __restrict__ O) {
  __shared__ __align__(16) __bf16 sK[2][64 * 72];   // K chunk [kcol][d]
  __shared__ __align__(16) __bf16 sVt[64 * 72];     // V chunk transposed
  __shared__ __align__(16) __bf16 sP[4][16 * 72];   // per-wave P tile

  const int t    = threadIdx.x;
  const int lane = t & 31;
  const int wid  = t >> 5;
  const int nl   = lane & 15;
  const int hf   = lane >> 4;
  const int b    = blockIdx.z;
  const int h    = blockIdx.y;
  const int qrow = blockIdx.x * 64 + wid * 16;

  // Q fragments for this wave's 16 rows (live for whole loop).
  Frag qa[2];
  {
    const __bf16* qp = Q + ((size_t)(b * kS + qrow + nl)) * kE + h * kD;
    qa[0].u[0] = *(const u32x4*)(qp + hf * 8);
    qa[0].u[1] = *(const u32x4*)(qp + 16 + hf * 8);
    qa[1].u[0] = *(const u32x4*)(qp + 32 + hf * 8);
    qa[1].u[1] = *(const u32x4*)(qp + 48 + hf * 8);
  }

  float m[8], l[8], alpha[8];
  f32x8 acc[4];
#pragma unroll
  for (int i = 0; i < 8; ++i) { m[i] = -INFINITY; l[i] = 0.0f; }
#pragma unroll
  for (int dt = 0; dt < 4; ++dt) acc[dt] = fzero8();

  const __bf16* Kb0 = K + ((size_t)(b * kS)) * kE + h * kD;
  const __bf16* Vb0 = V + ((size_t)(b * kS)) * kE + h * kD;

  const int nkc = kS / 64;
  if (wid == 0) {  // prologue: K tile 0
    tdm_load_tile64(Kb0, &sK[0][0], 64, (unsigned)kE);
  }

  for (int it = 0; it < nkc; ++it) {
    const int cur = it & 1;
    const __bf16* Vbase = Vb0 + (size_t)(it * 64) * kE;

    __syncthreads();  // previous iteration's sK/sVt/sP reads done
    // Stage V chunk transposed: thread handles 2 rows x 16 d values.
    {
      int rbase = t & 31, dg = t >> 5;
#pragma unroll
      for (int r2 = 0; r2 < 64; r2 += 32) {
        int row = rbase + r2;
        Frag vv;
        const __bf16* vr = &Vbase[(size_t)row * kE + dg * 16];
        vv.u[0] = *(const u32x4*)vr;
        vv.u[1] = *(const u32x4*)(vr + 8);
#pragma unroll
        for (int j = 0; j < 16; ++j)
          sVt[(dg * 16 + j) * 72 + row] = vv.v[j];
      }
    }
    if (wid == 0) {
      if (it + 1 < nkc) {
        tdm_load_tile64(Kb0 + (size_t)((it + 1) * 64) * kE, &sK[cur ^ 1][0], 64,
                        (unsigned)kE);
        __builtin_amdgcn_s_wait_tensorcnt(1);  // K tile it complete
      } else {
        __builtin_amdgcn_s_wait_tensorcnt(0);
      }
    }
    if (it + 1 < nkc) {  // prefetch next V chunk -> global_prefetch_b8
      __builtin_prefetch(Vbase + (size_t)64 * kE + (size_t)t * 64, 0, 1);
    }
    __syncthreads();

    // Scores: S = Q (16xD) * K^T (Dx64): 4 N-tiles x 2 K-steps = 8 WMMAs.
    const __bf16* cK = &sK[cur][0];
    f32x8 s[4];
#pragma unroll
    for (int tn = 0; tn < 4; ++tn) s[tn] = fzero8();
#pragma unroll
    for (int tn = 0; tn < 4; ++tn) {
      Frag kb0, kb1;
      const __bf16* p = &cK[(tn * 16 + nl) * 72 + hf * 8];
      kb0.u[0] = *(const u32x4*)p;        kb0.u[1] = *(const u32x4*)(p + 16);
      kb1.u[0] = *(const u32x4*)(p + 32); kb1.u[1] = *(const u32x4*)(p + 48);
      s[tn] = wmma_bf16f32(qa[0], kb0, s[tn]);
      s[tn] = wmma_bf16f32(qa[1], kb1, s[tn]);
    }

    // Online softmax in exp2 domain; row = i + 8*hf across 16 lanes of half.
#pragma unroll
    for (int i = 0; i < 8; ++i) {
      float mx = fmaxf(fmaxf(s[0][i], s[1][i]), fmaxf(s[2][i], s[3][i]));
      mx = red_max16(mx);
      float mn = fmaxf(m[i], mx);
      float al = exp2f(m[i] - mn);
      float p0 = exp2f(s[0][i] - mn);
      float p1 = exp2f(s[1][i] - mn);
      float p2 = exp2f(s[2][i] - mn);
      float p3 = exp2f(s[3][i] - mn);
      float rs = red_sum16((p0 + p1) + (p2 + p3));
      l[i]     = l[i] * al + rs;
      m[i]     = mn;
      alpha[i] = al;
      __bf16* pp = &sP[wid][(i + 8 * hf) * 72 + nl];
      pp[0]  = (__bf16)p0;
      pp[16] = (__bf16)p1;
      pp[32] = (__bf16)p2;
      pp[48] = (__bf16)p3;
    }
    __syncthreads();  // C-layout -> A-layout round trip through LDS

    // P*V: A = P (16x64, 2 K-steps), B = Vt (32x16 per d-tile): 8 WMMAs.
    Frag pa0, pa1;
    {
      const __bf16* pp = &sP[wid][nl * 72 + hf * 8];
      pa0.u[0] = *(const u32x4*)pp;        pa0.u[1] = *(const u32x4*)(pp + 16);
      pa1.u[0] = *(const u32x4*)(pp + 32); pa1.u[1] = *(const u32x4*)(pp + 48);
    }
#pragma unroll
    for (int dt = 0; dt < 4; ++dt) {
      Frag vb0, vb1;
      const __bf16* pv = &sVt[(dt * 16 + nl) * 72 + hf * 8];
      vb0.u[0] = *(const u32x4*)pv;        vb0.u[1] = *(const u32x4*)(pv + 16);
      vb1.u[0] = *(const u32x4*)(pv + 32); vb1.u[1] = *(const u32x4*)(pv + 48);
#pragma unroll
      for (int i = 0; i < 8; ++i) acc[dt][i] *= alpha[i];
      acc[dt] = wmma_bf16f32(pa0, vb0, acc[dt]);
      acc[dt] = wmma_bf16f32(pa1, vb1, acc[dt]);
    }
  }

  // Normalize and write ctx in [B, S, E] layout (heads interleaved).
  float inv[8];
#pragma unroll
  for (int i = 0; i < 8; ++i) inv[i] = 1.0f / l[i];
#pragma unroll
  for (int dt = 0; dt < 4; ++dt)
#pragma unroll
    for (int i = 0; i < 8; ++i) {
      int row = qrow + i + 8 * hf;
      int col = h * kD + dt * 16 + nl;
      O[((size_t)(b * kS + row)) * kE + col] = (__bf16)(acc[dt][i] * inv[i]);
    }
}

// ---------------------------------------------------------------------------
// Host launcher
// ---------------------------------------------------------------------------
extern "C" void kernel_launch(void* const* d_in, const int* in_sizes, int n_in,
                              void* d_out, int out_size, void* d_ws, size_t ws_size,
                              hipStream_t stream) {
  (void)in_sizes; (void)n_in; (void)out_size; (void)ws_size;
  const float* x  = (const float*)d_in[0];
  const float* Wq = (const float*)d_in[1];
  const float* Wk = (const float*)d_in[2];
  const float* Wv = (const float*)d_in[3];
  const float* Wo = (const float*)d_in[4];
  const float* bo = (const float*)d_in[5];
  float* out = (float*)d_out;

  // Workspace layout (bf16 elements), ~92 MB total.
  __bf16* ws = (__bf16*)d_ws;
  size_t o = 0;
  __bf16* xb  = ws + o; o += (size_t)kM * kE;
  __bf16* wqb = ws + o; o += (size_t)kE * kE;
  __bf16* wkb = ws + o; o += (size_t)kE * kE;
  __bf16* wvb = ws + o; o += (size_t)kE * kE;
  __bf16* wob = ws + o; o += (size_t)kE * kE;
  __bf16* qb  = ws + o; o += (size_t)kM * kE;
  __bf16* kb  = ws + o; o += (size_t)kM * kE;
  __bf16* vb  = ws + o; o += (size_t)kM * kE;
  __bf16* cxb = ws + o; o += (size_t)kM * kE;

  // Softmax scale (1/sqrt(D)) * log2(e), folded into Wq so Q*K^T produces
  // exp2-domain logits directly.
  const float kscale = 0.125f * 1.4426950408889634f;

  // 1) Casts to bf16.
  cast_f32_to_bf16<<<(kM * kE) / 1024, 256, 0, stream>>>(x,  xb,  kM * kE, 1.0f);
  cast_f32_to_bf16<<<(kE * kE) / 1024, 256, 0, stream>>>(Wq, wqb, kE * kE, kscale);
  cast_f32_to_bf16<<<(kE * kE) / 1024, 256, 0, stream>>>(Wk, wkb, kE * kE, 1.0f);
  cast_f32_to_bf16<<<(kE * kE) / 1024, 256, 0, stream>>>(Wv, wvb, kE * kE, 1.0f);
  cast_f32_to_bf16<<<(kE * kE) / 1024, 256, 0, stream>>>(Wo, wob, kE * kE, 1.0f);

  // 2) Q/K/V projections: [8192,1024] @ W^T.
  dim3 ggrid(kE / 128, kM / 128);
  gemm_xwT<true><<<ggrid, 256, 0, stream>>>(xb, wqb, qb, nullptr, nullptr, kE, kE);
  gemm_xwT<true><<<ggrid, 256, 0, stream>>>(xb, wkb, kb, nullptr, nullptr, kE, kE);
  gemm_xwT<true><<<ggrid, 256, 0, stream>>>(xb, wvb, vb, nullptr, nullptr, kE, kE);

  // 3) Flash attention per (batch, head, 64-query tile).
  flash_attn<<<dim3(kS / 64, kH, kB), 128, 0, stream>>>(qb, kb, vb, cxb);

  // 4) Output projection + bias (fp32 out).
  gemm_xwT<false><<<ggrid, 256, 0, stream>>>(cxb, wob, nullptr, out, bo, kE, kE);
}
